// Atten_Block_18519898981073
// MI455X (gfx1250) — matrix operand order repaired and checked by reading, not compiled, not run on hardware
//
#include <hip/hip_runtime.h>

#define CIN 256
#define CH4 64
#define NB  8
#define NN  4096   // H*W = 64*64

typedef __attribute__((ext_vector_type(16))) __bf16 v16bf;
typedef __attribute__((ext_vector_type(8)))  float  v8f;
typedef __attribute__((ext_vector_type(4)))  unsigned int v4u;
typedef __attribute__((ext_vector_type(8)))  int v8i;
typedef __attribute__((ext_vector_type(4)))  int v4i;

union Frag32 { uint4 u[2]; v16bf v; };

__device__ __forceinline__ unsigned short f2bf(float f) {
  unsigned int u = __float_as_uint(f);
  u += 0x7fffu + ((u >> 16) & 1u);          // round-to-nearest-even
  return (unsigned short)(u >> 16);
}

__device__ __forceinline__ v8f wmma_bf16(v16bf a, v16bf b, v8f c) {
  // (neg_a, A, neg_b, B, c_mod, C, reuse_a, reuse_b)
  return __builtin_amdgcn_wmma_f32_16x16x32_bf16(false, a, false, b, (short)0, c,
                                                 false, false);
}

// TDM: 1-D contiguous tile copy global -> LDS, nbytes must be 4096 here
// (tile_dim0 = 2048 elements of 2 bytes).  Descriptor per ISA 08_async_tensor
// 8.3/8.4: group0 = {count=1, lds_addr, global_addr, type=2}, group1 packs
// data_size=1 (2B), tensor_dim0 = tile_dim0 = 2048, dims 1/2 unused.
__device__ __forceinline__ void tdm_load_k(const unsigned short* gsrc,
                                           unsigned int lds_off) {
  unsigned long long ga = (unsigned long long)(uintptr_t)gsrc;
  v4u g0;
  g0[0] = 1u;                                                   // count=1
  g0[1] = lds_off;                                              // LDS byte addr
  g0[2] = (unsigned int)ga;                                     // gaddr lo
  g0[3] = (unsigned int)((ga >> 32) & 0x1FFFFFFu) | (2u << 30); // gaddr hi|type=2
  v8i g1;
  g1[0] = (1 << 16);                              // wg_mask=0, data_size=1 (2B)
  g1[1] = (int)((2048u & 0xFFFFu) << 16);         // tensor_dim0[15:0] at bit48
  g1[2] = (int)((2048u >> 16) | (1u << 16));      // tensor_dim0 hi | tensor_dim1=1
  g1[3] = (int)(2048u << 16);                     // tensor_dim1 hi | tile_dim0=2048
  g1[4] = 0;                                      // tile_dim1=0, tile_dim2=0
  g1[5] = 2048;                                   // tensor_dim0_stride lo
  g1[6] = 0;
  g1[7] = 0;
  v4i gz = {0, 0, 0, 0};
#if __clang_major__ >= 23
  v8i gz8 = {0, 0, 0, 0, 0, 0, 0, 0};
  __builtin_amdgcn_tensor_load_to_lds(g0, g1, gz, gz, gz8, 0);
#else
  __builtin_amdgcn_tensor_load_to_lds(g0, g1, gz, gz, 0);
#endif
}

// ---------------------------------------------------------------------------
// Kernel 0: pre-pack all weights to bf16.
// ---------------------------------------------------------------------------
__global__ __launch_bounds__(256) void conv_weights(
    const float* __restrict__ W1, const float* __restrict__ W2,
    const float* __restrict__ W3, const float* __restrict__ W4,
    unsigned short* __restrict__ wq, unsigned short* __restrict__ wk,
    unsigned short* __restrict__ wv, unsigned short* __restrict__ w4b) {
  int i = blockIdx.x * blockDim.x + threadIdx.x;   // 0..65535
  int which = i >> 14;
  int j = i & 16383;
  if (which == 0)      wq[j]  = f2bf(W1[j]);
  else if (which == 1) wk[j]  = f2bf(W2[j]);
  else if (which == 2) wv[j]  = f2bf(W3[j]);
  else                 w4b[j] = f2bf(W4[j]);
}

// ---------------------------------------------------------------------------
// Kernel 1: theta/phi/g = W{1,2,3} @ x + b  -> bf16 [B, N, 64] each.
// ---------------------------------------------------------------------------
__global__ __launch_bounds__(256) void qkv_proj(
    const float* __restrict__ x,
    const float* __restrict__ b1, const float* __restrict__ b2,
    const float* __restrict__ b3,
    const unsigned short* __restrict__ wq, const unsigned short* __restrict__ wk,
    const unsigned short* __restrict__ wv,
    unsigned short* __restrict__ Qb, unsigned short* __restrict__ Kb,
    unsigned short* __restrict__ Vb) {
  __shared__ __align__(16) unsigned short ldsx[8][16 * 32];  // per-wave [n][c] bf16

  const int lane   = threadIdx.x & 31;
  const int wave   = threadIdx.x >> 5;
  const int tile   = blockIdx.x * 8 + wave;      // 0..2047
  const int b      = tile / (NN / 16);
  const int n0     = (tile % (NN / 16)) * 16;
  const int lh     = lane & 15;
  const int half8  = (lane < 16) ? 0 : 8;
  const int half16 = (lane < 16) ? 0 : 16;

  const unsigned short* wptr[3] = {wq, wk, wv};
  const float*          bptr[3] = {b1, b2, b3};

  v8f acc[3][4];
#pragma unroll
  for (int w = 0; w < 3; ++w)
#pragma unroll
    for (int t = 0; t < 4; ++t)
#pragma unroll
      for (int r = 0; r < 8; ++r)
        acc[w][t][r] = bptr[w][t * 16 + r + half8];   // bias in C accumulator

  const float* xrow = x + (size_t)b * CIN * NN + n0;  // + c*NN
  unsigned short* myx = &ldsx[wave][0];

  for (int c0 = 0; c0 < CIN; c0 += 32) {
    // stage x[c0+lane][n0..n0+15] -> lds [n][c=lane] as bf16 (wave-private)
    const float4* src = (const float4*)(xrow + (size_t)(c0 + lane) * NN);
    float xv[16];
#pragma unroll
    for (int q = 0; q < 4; ++q) {
      float4 f = src[q];
      xv[q * 4 + 0] = f.x; xv[q * 4 + 1] = f.y;
      xv[q * 4 + 2] = f.z; xv[q * 4 + 3] = f.w;
    }
#pragma unroll
    for (int n = 0; n < 16; ++n) myx[n * 32 + lane] = f2bf(xv[n]);

    // B fragment: col n = lh, 16 contiguous K (=c) per lane, split by half-wave
    Frag32 bf;
    const uint4* bl = (const uint4*)(myx + lh * 32 + half16);
    bf.u[0] = bl[0];
    bf.u[1] = bl[1];

#pragma unroll
    for (int w = 0; w < 3; ++w) {
#pragma unroll
      for (int t = 0; t < 4; ++t) {
        Frag32 af;  // A: row m = t*16+lh, 16 contiguous K {half16..half16+15}
        const unsigned short* wrow = wptr[w] + (t * 16 + lh) * CIN + c0;
        af.u[0] = *(const uint4*)(wrow + half16);
        af.u[1] = *(const uint4*)(wrow + half16 + 8);
        acc[w][t] = wmma_bf16(af.v, bf.v, acc[w][t]);
      }
    }
  }

  unsigned short* outp[3] = {Qb, Kb, Vb};
#pragma unroll
  for (int w = 0; w < 3; ++w) {
    unsigned short* dst = outp[w] + ((size_t)(b * NN + n0 + lh)) * CH4;
#pragma unroll
    for (int t = 0; t < 4; ++t)
#pragma unroll
      for (int r = 0; r < 8; ++r)
        dst[t * 16 + r + half8] = f2bf(acc[w][t][r]);
  }
}

// ---------------------------------------------------------------------------
// Kernel 2: flash attention.  Each wave: 16 query rows, d=64.
// K tile staged by the Tensor Data Mover (contiguous 4KB), V tile transposed
// into LDS by vector loads + ds scatter, next V rows prefetched.
// ---------------------------------------------------------------------------
__global__ __launch_bounds__(256) void attention(
    const unsigned short* __restrict__ Qb,
    const unsigned short* __restrict__ Kb,
    const unsigned short* __restrict__ Vb,
    unsigned short* __restrict__ Og) {
  __shared__ __align__(16) unsigned short ldsK[32 * 64];      // [j][c]
  __shared__ __align__(16) unsigned short ldsV[64 * 32];      // [c][j] (transposed)
  __shared__ __align__(16) unsigned short ldsP[8][16 * 32];   // per-wave [i][j]

  const int lane   = threadIdx.x & 31;
  const int wave   = threadIdx.x >> 5;
  const int bpb    = NN / 128;                    // i-blocks per batch
  const int b      = blockIdx.x / bpb;
  const int i0     = (blockIdx.x % bpb) * 128 + wave * 16;
  const int lh     = lane & 15;
  const int half8  = (lane < 16) ? 0 : 8;
  const int half16 = (lane < 16) ? 0 : 16;

  // Q fragments, held for the whole j loop (rows i0+lh, K = c in two 32-chunks)
  Frag32 qa[2];
  const unsigned short* qrow = Qb + ((size_t)(b * NN + i0 + lh)) * CH4;
#pragma unroll
  for (int f = 0; f < 2; ++f) {
    qa[f].u[0] = *(const uint4*)(qrow + f * 32 + half16);
    qa[f].u[1] = *(const uint4*)(qrow + f * 32 + half16 + 8);
  }

  float m[8], l[8];
  v8f o[4];
#pragma unroll
  for (int r = 0; r < 8; ++r) { m[r] = -1e30f; l[r] = 0.f; }
#pragma unroll
  for (int t = 0; t < 4; ++t)
#pragma unroll
    for (int r = 0; r < 8; ++r) o[t][r] = 0.f;

  const int sj = threadIdx.x >> 3;        // 0..31 : j row staged by this thread
  const int sc = (threadIdx.x & 7) * 8;   // 0..56 : c chunk staged by this thread
  const unsigned int ldsK_off = (unsigned int)(uintptr_t)&ldsK[0];

  for (int j0 = 0; j0 < NN; j0 += 32) {
    __syncthreads();   // previous iteration's LDS reads done

    // K tile: one TDM descriptor, issued by wave 0 only
    if (wave == 0)
      tdm_load_k(Kb + ((size_t)(b * NN + j0)) * CH4, ldsK_off);

    // V tile: transpose into [c][j] via vector load + ds scatter
    {
      size_t rowoff = ((size_t)(b * NN + j0 + sj)) * CH4 + sc;
      union { uint4 u; unsigned short s[8]; } vv;
      vv.u = *(const uint4*)(Vb + rowoff);
#pragma unroll
      for (int e = 0; e < 8; ++e) ldsV[(sc + e) * 32 + sj] = vv.s[e];
      if (j0 + 32 < NN)
        __builtin_prefetch(Vb + rowoff + 32 * CH4, 0, 1);  // next j-block rows
    }

    if (wave == 0) __builtin_amdgcn_s_wait_tensorcnt(0);
    __syncthreads();

    // S tiles (two 16-col halves of the 32-wide j block), K-dim = 64 = 2 WMMAs
    v8f s[2];
#pragma unroll
    for (int jh = 0; jh < 2; ++jh) {
      v8f a;
#pragma unroll
      for (int r = 0; r < 8; ++r) a[r] = 0.f;
      const int jl = jh * 16 + lh;
#pragma unroll
      for (int f = 0; f < 2; ++f) {
        Frag32 kf;
        const uint4* kp = (const uint4*)(&ldsK[jl * 64 + f * 32 + half16]);
        kf.u[0] = kp[0];
        kf.u[1] = kp[1];
        a = wmma_bf16(qa[f].v, kf.v, a);
      }
      s[jh] = a;
    }

    // online softmax: row stats live per-lane (8 rows = D-layout vgpr index)
#pragma unroll
    for (int r = 0; r < 8; ++r) {
      float mx = fmaxf(s[0][r], s[1][r]);
      mx = fmaxf(mx, __shfl_xor(mx, 1));
      mx = fmaxf(mx, __shfl_xor(mx, 2));
      mx = fmaxf(mx, __shfl_xor(mx, 4));
      mx = fmaxf(mx, __shfl_xor(mx, 8));
      float mn    = fmaxf(m[r], mx);
      float alpha = __expf(m[r] - mn);
      float p0    = __expf(s[0][r] - mn);
      float p1    = __expf(s[1][r] - mn);
      ldsP[wave][(r + half8) * 32 + lh]      = f2bf(p0);
      ldsP[wave][(r + half8) * 32 + 16 + lh] = f2bf(p1);
      float ps = p0 + p1;
      ps += __shfl_xor(ps, 1);
      ps += __shfl_xor(ps, 2);
      ps += __shfl_xor(ps, 4);
      ps += __shfl_xor(ps, 8);
      l[r] = l[r] * alpha + ps;
      m[r] = mn;
#pragma unroll
      for (int t = 0; t < 4; ++t) o[t][r] *= alpha;
    }

    // re-read P in A layout (wave-private LDS; ds ops are in-order per wave)
    Frag32 pf;
    pf.u[0] = *(const uint4*)(&ldsP[wave][lh * 32 + half16]);
    pf.u[1] = *(const uint4*)(&ldsP[wave][lh * 32 + half16 + 8]);

    // O += P @ V : 4 c-tiles of 16
#pragma unroll
    for (int t = 0; t < 4; ++t) {
      Frag32 vf;
      const uint4* vp = (const uint4*)(&ldsV[(t * 16 + lh) * 32 + half16]);
      vf.u[0] = vp[0];
      vf.u[1] = vp[1];
      o[t] = wmma_bf16(pf.v, vf.v, o[t]);
    }
  }

  // normalize and store O -> bf16 [B, N, 64]
#pragma unroll
  for (int r = 0; r < 8; ++r) {
    float inv   = 1.f / l[r];
    size_t base = ((size_t)(b * NN + i0 + r + half8)) * CH4 + lh;
#pragma unroll
    for (int t = 0; t < 4; ++t) Og[base + t * 16] = f2bf(o[t][r] * inv);
  }
}

// ---------------------------------------------------------------------------
// Kernel 3: out = x + b4 + W4 @ O^T  (16 co-tiles x 2 WMMA per 16-col n-tile)
// ---------------------------------------------------------------------------
__global__ __launch_bounds__(256) void out_proj(
    const float* __restrict__ x, const float* __restrict__ b4,
    const unsigned short* __restrict__ w4b, const unsigned short* __restrict__ Og,
    float* __restrict__ out) {
  const int lane   = threadIdx.x & 31;
  const int wave   = threadIdx.x >> 5;
  const int tile   = blockIdx.x * 8 + wave;
  const int b      = tile / (NN / 16);
  const int n0     = (tile % (NN / 16)) * 16;
  const int lh     = lane & 15;
  const int half8  = (lane < 16) ? 0 : 8;
  const int half16 = (lane < 16) ? 0 : 16;

  // B fragments from Og (col n = n0+lh, K = c4), reused across all 16 co-tiles
  Frag32 gf[2];
  const unsigned short* grow = Og + ((size_t)(b * NN + n0 + lh)) * CH4;
#pragma unroll
  for (int f = 0; f < 2; ++f) {
    gf[f].u[0] = *(const uint4*)(grow + f * 32 + half16);
    gf[f].u[1] = *(const uint4*)(grow + f * 32 + half16 + 8);
  }

#pragma unroll 4
  for (int t = 0; t < 16; ++t) {
    v8f acc;
#pragma unroll
    for (int r = 0; r < 8; ++r) acc[r] = b4[t * 16 + r + half8];
#pragma unroll
    for (int f = 0; f < 2; ++f) {
      Frag32 af;
      const unsigned short* wrow = w4b + (t * 16 + lh) * CH4 + f * 32;
      af.u[0] = *(const uint4*)(wrow + half16);
      af.u[1] = *(const uint4*)(wrow + half16 + 8);
      acc = wmma_bf16(af.v, gf[f].v, acc);
    }
#pragma unroll
    for (int r = 0; r < 8; ++r) {
      int co     = t * 16 + r + half8;
      size_t idx = ((size_t)(b * CIN + co)) * NN + n0 + lh;
      out[idx]   = x[idx] + acc[r];   // residual
    }
  }
}

// ---------------------------------------------------------------------------
extern "C" void kernel_launch(void* const* d_in, const int* in_sizes, int n_in,
                              void* d_out, int out_size, void* d_ws, size_t ws_size,
                              hipStream_t stream) {
  const float* x  = (const float*)d_in[0];
  const float* W1 = (const float*)d_in[1];
  const float* b1 = (const float*)d_in[2];
  const float* W2 = (const float*)d_in[3];
  const float* b2 = (const float*)d_in[4];
  const float* W3 = (const float*)d_in[5];
  const float* b3 = (const float*)d_in[6];
  const float* W4 = (const float*)d_in[7];
  const float* b4 = (const float*)d_in[8];
  float* out = (float*)d_out;

  // workspace layout (bf16), ~16.2 MB total
  unsigned short* wq  = (unsigned short*)d_ws;          // 64*256
  unsigned short* wk  = wq + 64 * 256;
  unsigned short* wv  = wk + 64 * 256;
  unsigned short* w4b = wv + 64 * 256;                  // 256*64
  unsigned short* Qb  = w4b + 256 * 64;                 // B*N*64 each
  unsigned short* Kb  = Qb + (size_t)NB * NN * CH4;
  unsigned short* Vb  = Kb + (size_t)NB * NN * CH4;
  unsigned short* Og  = Vb + (size_t)NB * NN * CH4;

  conv_weights<<<256, 256, 0, stream>>>(W1, W2, W3, W4, wq, wk, wv, w4b);
  qkv_proj<<<256, 256, 0, stream>>>(x, b1, b2, b3, wq, wk, wv, Qb, Kb, Vb);
  attention<<<NB * (NN / 128), 256, 0, stream>>>(Qb, Kb, Vb, Og);
  out_proj<<<256, 256, 0, stream>>>(x, b4, w4b, Og, out);
}
